// Attention_83623013253180
// MI455X (gfx1250) — matrix operand order repaired and checked by reading, not compiled, not run on hardware
//
#include <hip/hip_runtime.h>
#include <hip/hip_bf16.h>

typedef __attribute__((ext_vector_type(16))) __bf16 v16bf;
typedef __attribute__((ext_vector_type(8)))  __bf16 v8bf;
typedef __attribute__((ext_vector_type(4)))  __bf16 v4bf;
typedef __attribute__((ext_vector_type(8)))  float  v8f;
typedef __attribute__((ext_vector_type(4)))  float  v4f;
typedef __attribute__((ext_vector_type(4)))  unsigned int v4u;
typedef __attribute__((ext_vector_type(8)))  int    v8i;
typedef __attribute__((ext_vector_type(4)))  int    v4i;

#define WMMA_BF16(a, b, c) \
  __builtin_amdgcn_wmma_f32_16x16x32_bf16(false, (a), false, (b), (short)0, (c), false, false)

// Build a 16x32 A-fragment / 32x16 B-fragment half from two contiguous
// 8-element (16-byte) chunks in LDS/global.
__device__ __forceinline__ v16bf make_frag(const __bf16* lo, const __bf16* hi8) {
  v8bf a = *(const v8bf*)lo;
  v8bf b = *(const v8bf*)hi8;
  v16bf r;
#pragma unroll
  for (int i = 0; i < 8; ++i) { r[i] = a[i]; r[i + 8] = b[i]; }
  return r;
}

// ---------------------------------------------------------------------------
// Tensor Data Mover: DMA a 2D bf16 tile (rows x 128 elems) from global memory
// into LDS, padding each 256B row by 16B so the LDS image has a 272B row
// stride (matches Ks[64][136], bank-conflict friendly).
// Descriptor layout per CDNA5 ISA.
// Issued by one wave; tracked with TENSORcnt.
// ---------------------------------------------------------------------------
__device__ __forceinline__ void tdm_load_k_tile(const __bf16* gsrc,
                                                unsigned lds_byte_off,
                                                int rows, int row_stride_elems) {
  const unsigned long long ga = (unsigned long long)(uintptr_t)gsrc;
  v4u g0;
  g0[0] = 1u;                                   // count=1, user descriptor
  g0[1] = lds_byte_off;                         // lds_addr
  g0[2] = (unsigned)ga;                         // global_addr[31:0]
  g0[3] = ((unsigned)(ga >> 32) & 0x01FFFFFFu)  // global_addr[56:32]
          | (2u << 30);                         // type=2 ("image")
  v8i g1;
  g1[0] = (1 << 16)      // data_size = 2 bytes
        | (1 << 20)      // pad_enable
        | (5 << 22)      // pad_interval: every 64 DWORDs (256B)
        | (3 << 25);     // pad_amount: 4 DWORDs (16B)
  g1[1] = (128 & 0xFFFF) << 16;                 // tensor_dim0 = 128 (bits 79:48)
  g1[2] = ((rows & 0xFFFF) << 16);              // tensor_dim0 hi=0 | tensor_dim1 lo
  g1[3] = (128 << 16);                          // tensor_dim1 hi=0 | tile_dim0 = 128
  g1[4] = rows & 0xFFFF;                        // tile_dim1 = rows, tile_dim2 = 0
  g1[5] = row_stride_elems;                     // tensor_dim0_stride (elems), low 32
  g1[6] = 0;
  g1[7] = 0;
  v4i z4; z4[0] = 0; z4[1] = 0; z4[2] = 0; z4[3] = 0;
  v8i z8; 
#pragma unroll
  for (int i = 0; i < 8; ++i) z8[i] = 0;
  // 6-arg form (upstream clang / therock headers): groups 0..3 + extra + cpol.
  __builtin_amdgcn_tensor_load_to_lds(g0, g1, z4, z4, z8, 0);
}

// ---------------------------------------------------------------------------
// Tiled GEMM:  C[M,N] = (A[M,K] @ B[K,N] + bias) * scale
// A: fp32 or bf16 (converted to bf16 in LDS), B: fp32 (converted), C: OT.
// Block tile 128x128, K-step 32, 256 threads = 8 waves in a 4(M) x 2(N) grid,
// each wave computes a 32x64 tile = 2x4 WMMA accumulators.
// Global->register double buffering overlaps VMEM with the WMMA stream.
// ---------------------------------------------------------------------------
template <typename AT, typename OT>
__global__ __launch_bounds__(256) void gemm_wmma_bf16(
    const AT* __restrict__ A, const float* __restrict__ Bm,
    const float* __restrict__ bias, const float* __restrict__ scale_p,
    OT* __restrict__ C, int M, int N, int K) {
  __shared__ __bf16 As[128][40];   // 128x32 tile, row stride 40 (80B, 16B-aligned)
  __shared__ __bf16 Bt[128][40];   // B tile stored transposed: [N][K]

  const int t = threadIdx.x;
  const int lane = t & 31;
  const int wave = t >> 5;
  const int wm = wave & 3;         // 0..3 : wave row  (32 rows each)
  const int wn = wave >> 2;        // 0..1 : wave col  (64 cols each)
  const int l15 = lane & 15;
  const int hi = lane >> 4;
  const int bm = blockIdx.y * 128;
  const int bn = blockIdx.x * 128;

  v8f acc[2][4];
#pragma unroll
  for (int i = 0; i < 2; ++i)
#pragma unroll
    for (int j = 0; j < 4; ++j)
#pragma unroll
      for (int r = 0; r < 8; ++r) acc[i][j][r] = 0.0f;

  AT    areg[16];
  float breg[16];
  const int ra = t >> 3;               // 0..31  (A: 4 passes of 32 rows)
  const int ca = (t & 7) * 4;          // 0..28
  const int rb = t >> 5;               // 0..7   (B: 4 passes of 8 rows)
  const int cb = (t & 31) * 4;         // 0..124

  auto g_load = [&](int k0) {
#pragma unroll
    for (int p = 0; p < 4; ++p) {
      const AT* src = A + (size_t)(bm + ra + p * 32) * K + k0 + ca;
      if constexpr (sizeof(AT) == 4) {
        v4f v = *(const v4f*)src;
#pragma unroll
        for (int e = 0; e < 4; ++e) areg[p * 4 + e] = v[e];
      } else {
        v4bf v = *(const v4bf*)src;
#pragma unroll
        for (int e = 0; e < 4; ++e) areg[p * 4 + e] = v[e];
      }
    }
#pragma unroll
    for (int p = 0; p < 4; ++p) {
      v4f v = *(const v4f*)(Bm + (size_t)(k0 + rb + p * 8) * N + bn + cb);
#pragma unroll
      for (int e = 0; e < 4; ++e) breg[p * 4 + e] = v[e];
    }
  };
  auto lds_store = [&]() {
#pragma unroll
    for (int p = 0; p < 4; ++p) {
      v4bf o;
#pragma unroll
      for (int e = 0; e < 4; ++e) o[e] = (__bf16)(float)areg[p * 4 + e];
      *(v4bf*)&As[ra + p * 32][ca] = o;
    }
#pragma unroll
    for (int p = 0; p < 4; ++p)
#pragma unroll
      for (int e = 0; e < 4; ++e)
        Bt[cb + e][rb + p * 8] = (__bf16)breg[p * 4 + e];
  };

  g_load(0);
  for (int k0 = 0; k0 < K; k0 += 32) {
    lds_store();
    __syncthreads();
    if (k0 + 32 < K) g_load(k0 + 32);   // prefetch next tile into registers

    v16bf afr[2], bfr[4];
#pragma unroll
    for (int i = 0; i < 2; ++i) {
      const __bf16* base = &As[wm * 32 + i * 16 + l15][0];
      afr[i] = make_frag(base + hi * 8, base + 16 + hi * 8);
    }
#pragma unroll
    for (int j = 0; j < 4; ++j) {
      const __bf16* base = &Bt[wn * 64 + j * 16 + l15][0];
      bfr[j] = make_frag(base + 8 * hi, base + 16 + 8 * hi);
    }
#pragma unroll
    for (int i = 0; i < 2; ++i)
#pragma unroll
      for (int j = 0; j < 4; ++j)
        acc[i][j] = WMMA_BF16(afr[i], bfr[j], acc[i][j]);
    __syncthreads();
  }

  const float scl = scale_p ? scale_p[0] : 1.0f;
#pragma unroll
  for (int i = 0; i < 2; ++i) {
#pragma unroll
    for (int j = 0; j < 4; ++j) {
      const int cg = bn + wn * 64 + j * 16 + l15;
      const float bv = bias ? bias[cg] : 0.0f;
#pragma unroll
      for (int r = 0; r < 8; ++r) {
        const int rg = bm + wm * 32 + i * 16 + r + 8 * hi;
        C[(size_t)rg * N + cg] = (OT)((acc[i][j][r] + bv) * scl);
      }
    }
  }
}

// ---------------------------------------------------------------------------
// Flash attention (causal), bf16 compute / fp32 softmax + accumulators.
// One workgroup = (batch b, head h, 128 query rows); 8 waves x 16 rows each.
// KV tiles of 64 rows. DQK = DV = 128.
// K tile staged by the Tensor Data Mover (wave 0 issues, TENSORcnt waited);
// V tile staged transposed by all waves with register prefetch of next tile.
// ---------------------------------------------------------------------------
__global__ __launch_bounds__(256) void flash_attn_kernel(
    const __bf16* __restrict__ qb, const __bf16* __restrict__ kb,
    const __bf16* __restrict__ vb, __bf16* __restrict__ ctx,
    int L, int H, int G, int HPG) {
  __shared__ __bf16 Ks[64][136];       // K tile row-major: IS the B^T layout for Q@K^T
  __shared__ __bf16 Vt[128][72];       // V tile transposed: [dv][kv]
  __shared__ __bf16 Ps[8][16][72];     // per-wave P staging (D-layout -> A-layout)

  const int t = threadIdx.x;
  const int lane = t & 31;
  const int wave = t >> 5;
  const int l15 = lane & 15;
  const int hi = lane >> 4;
  const int qtile = blockIdx.x;
  const int h = blockIdx.y;
  const int b = blockIdx.z;
  const int g = h / HPG;
  const int qrow0 = qtile * 128 + wave * 16;
  const float sm_scale = 0.08838834764831845f;  // 1/sqrt(128)

  // Q fragments for this wave's 16 rows (4 K-steps of 32 over DQK=128).
  v16bf qfr[4];
  {
    const __bf16* qrow = qb + ((size_t)(b * L + qrow0 + l15) * H + h) * 128;
#pragma unroll
    for (int ks = 0; ks < 4; ++ks)
      qfr[ks] = make_frag(qrow + ks * 32 + hi * 8, qrow + ks * 32 + 16 + hi * 8);
  }

  v8f acc[8];
#pragma unroll
  for (int dt = 0; dt < 8; ++dt)
#pragma unroll
    for (int r = 0; r < 8; ++r) acc[dt][r] = 0.0f;
  float mrow[8], lrow[8];
#pragma unroll
  for (int r = 0; r < 8; ++r) { mrow[r] = -3.0e38f; lrow[r] = 0.0f; }

  const int rv = t >> 4;               // 0..15
  const int cv = (t & 15) * 8;         // 0..120
  v8bf vreg[4];
  auto v_gload = [&](int kv0) {
#pragma unroll
    for (int p = 0; p < 4; ++p)
      vreg[p] = *(const v8bf*)(vb +
          ((size_t)(b * L + kv0 + rv + p * 16) * G + g) * 128 + cv);
  };
  const unsigned ks_lds_off = (unsigned)(uintptr_t)&Ks[0][0];

  const int ntiles = (qtile + 1) * 2;  // causal: kv up to end of this q tile
  v_gload(0);
  for (int j = 0; j < ntiles; ++j) {
    const int kv0 = j * 64;
    // ---- K tile: Tensor Data Mover, one DMA for the whole 64x128 tile ----
    if (wave == 0)
      tdm_load_k_tile(kb + ((size_t)(b * L + kv0) * G + g) * 128,
                      ks_lds_off, 64, G * 128);
    // ---- V tile: store prefetched registers transposed into LDS ----
#pragma unroll
    for (int p = 0; p < 4; ++p)
#pragma unroll
      for (int e = 0; e < 8; ++e) Vt[cv + e][rv + p * 16] = vreg[p][e];
    if (j + 1 < ntiles) v_gload(kv0 + 64);   // prefetch next V tile
    if (wave == 0) __builtin_amdgcn_s_wait_tensorcnt(0);
    __syncthreads();

    if (kv0 <= qrow0 + 15) {           // tile not fully above diagonal for this wave
      // ---- S = Q @ K^T ----
      v8f s[4];
#pragma unroll
      for (int nt = 0; nt < 4; ++nt) {
#pragma unroll
        for (int r = 0; r < 8; ++r) s[nt][r] = 0.0f;
#pragma unroll
        for (int ks = 0; ks < 4; ++ks) {
          const __bf16* base = &Ks[nt * 16 + l15][ks * 32 + 8 * hi];
          v16bf bf = make_frag(base, base + 16);
          s[nt] = WMMA_BF16(qfr[ks], bf, s[nt]);
        }
      }
      // ---- scale + causal mask ----
      const bool needmask = (kv0 + 63 > qrow0);
#pragma unroll
      for (int nt = 0; nt < 4; ++nt) {
#pragma unroll
        for (int r = 0; r < 8; ++r) {
          float x = s[nt][r] * sm_scale;
          if (needmask) {
            const int rowg = qrow0 + r + 8 * hi;
            const int colg = kv0 + nt * 16 + l15;
            if (colg > rowg) x = -3.0e38f;
          }
          s[nt][r] = x;
        }
      }
      // ---- online softmax (row lives across 16 lanes of each half-wave) ----
      float alpha[8];
#pragma unroll
      for (int r = 0; r < 8; ++r) {
        float vmax = fmaxf(fmaxf(s[0][r], s[1][r]), fmaxf(s[2][r], s[3][r]));
        vmax = fmaxf(vmax, __shfl_xor(vmax, 1, 32));
        vmax = fmaxf(vmax, __shfl_xor(vmax, 2, 32));
        vmax = fmaxf(vmax, __shfl_xor(vmax, 4, 32));
        vmax = fmaxf(vmax, __shfl_xor(vmax, 8, 32));
        const float mnew = fmaxf(mrow[r], vmax);
        alpha[r] = __expf(mrow[r] - mnew);
        mrow[r] = mnew;
        float rs = 0.0f;
#pragma unroll
        for (int nt = 0; nt < 4; ++nt) {
          const float p = __expf(s[nt][r] - mnew);
          s[nt][r] = p;
          rs += p;
        }
        rs += __shfl_xor(rs, 1, 32);
        rs += __shfl_xor(rs, 2, 32);
        rs += __shfl_xor(rs, 4, 32);
        rs += __shfl_xor(rs, 8, 32);
        lrow[r] = lrow[r] * alpha[r] + rs;
      }
#pragma unroll
      for (int dt = 0; dt < 8; ++dt)
#pragma unroll
        for (int r = 0; r < 8; ++r) acc[dt][r] *= alpha[r];
      // ---- P: D-layout -> A-layout via per-wave LDS patch ----
#pragma unroll
      for (int nt = 0; nt < 4; ++nt)
#pragma unroll
        for (int r = 0; r < 8; ++r)
          Ps[wave][r + 8 * hi][nt * 16 + l15] = (__bf16)s[nt][r];
      // ---- ctx += P @ V ----
#pragma unroll
      for (int kp = 0; kp < 2; ++kp) {
        const __bf16* pb = &Ps[wave][l15][kp * 32 + hi * 8];
        v16bf pf = make_frag(pb, pb + 16);
#pragma unroll
        for (int dt = 0; dt < 8; ++dt) {
          const __bf16* vbase = &Vt[dt * 16 + l15][kp * 32 + 8 * hi];
          v16bf vf = make_frag(vbase, vbase + 16);
          acc[dt] = WMMA_BF16(pf, vf, acc[dt]);
        }
      }
    }
    __syncthreads();
  }

  // ---- normalize and store ctx (bf16) as [B, L, H*DV] ----
#pragma unroll
  for (int r = 0; r < 8; ++r) {
    const int rowg = qrow0 + r + 8 * hi;
    const float inv = 1.0f / lrow[r];
    __bf16* dst = ctx + ((size_t)(b * L + rowg) * H + h) * 128;
#pragma unroll
    for (int dt = 0; dt < 8; ++dt)
      dst[dt * 16 + l15] = (__bf16)(acc[dt][r] * inv);
  }
}

// ---------------------------------------------------------------------------
extern "C" void kernel_launch(void* const* d_in, const int* in_sizes, int n_in,
                              void* d_out, int out_size, void* d_ws,
                              size_t ws_size, hipStream_t stream) {
  const float* in_q = (const float*)d_in[0];
  const float* in_k = (const float*)d_in[1];
  const float* in_v = (const float*)d_in[2];
  const float* Wq   = (const float*)d_in[3];
  const float* bq   = (const float*)d_in[4];
  const float* Wk   = (const float*)d_in[5];
  const float* bk   = (const float*)d_in[6];
  const float* Wv   = (const float*)d_in[7];
  const float* bv   = (const float*)d_in[8];
  const float* Wo   = (const float*)d_in[9];
  const float* bo   = (const float*)d_in[10];
  const float* qsc  = (const float*)d_in[11];

  const int B = 2, L = 2048, D = 1024, H = 16, G = 2, DQK = 128;
  const int M = B * L;  // 4096 tokens

  __bf16* qb = (__bf16*)d_ws;                       // [M, H*DQK]  16 MB
  __bf16* kb = qb + (size_t)M * H * DQK;            // [M, G*DQK]   2 MB
  __bf16* vb = kb + (size_t)M * G * DQK;            // [M, G*DQK]   2 MB
  __bf16* cx = vb + (size_t)M * G * DQK;            // [M, H*DQK]  16 MB

  dim3 blk(256);
  // Q = (in_q @ Wq + bq) * q_extra_scale  -> bf16
  gemm_wmma_bf16<float, __bf16><<<dim3((H * DQK) / 128, M / 128), blk, 0, stream>>>(
      in_q, Wq, bq, qsc, qb, M, H * DQK, D);
  // K = in_k @ Wk + bk -> bf16
  gemm_wmma_bf16<float, __bf16><<<dim3((G * DQK) / 128, M / 128), blk, 0, stream>>>(
      in_k, Wk, bk, nullptr, kb, M, G * DQK, D);
  // V = in_v @ Wv + bv -> bf16
  gemm_wmma_bf16<float, __bf16><<<dim3((G * DQK) / 128, M / 128), blk, 0, stream>>>(
      in_v, Wv, bv, nullptr, vb, M, G * DQK, D);
  // causal flash attention -> ctx (bf16)
  flash_attn_kernel<<<dim3(L / 128, H, B), blk, 0, stream>>>(
      qb, kb, vb, cx, L, H, G, H / G);
  // out = ctx @ Wo + bo -> fp32
  gemm_wmma_bf16<__bf16, float><<<dim3(D / 128, M / 128), blk, 0, stream>>>(
      cx, Wo, bo, nullptr, (float*)d_out, M, D, H * DQK);
}
